// ElevationGuidedGate_67319317397649
// MI455X (gfx1250) — matrix operand order repaired and checked by reading, not compiled, not run on hardware
//
#include <hip/hip_runtime.h>
#include <math.h>

typedef __attribute__((ext_vector_type(16))) _Float16 v16h;
typedef __attribute__((ext_vector_type(8)))  float    v8f;

#define Bn   8
#define Hh   256
#define Ww   256
#define HWs  (Hh * Ww)        // 65536
#define Ls   HWs
#define Dd   128
#define NPIX (Bn * HWs)       // 524288 pixels (h path)
#define NROW (Bn * Ls)        // 524288 rows (x path)

// ---- workspace layout (floats) ----
// [0..8]    : sum of tap values          (9)
// [9..53]   : sum of tap cross-products  (45, upper triangle)
// [96..111] : BN scale per channel       (16)
// [112..127]: BN shift per channel       (16)
// [128]     : global sum of norm_dx      (1)
#define STATS_F   256
#define SCALE_I   96
#define SHIFT_I   112
#define NSUM_I    128
#define HREF_F    (Bn * HWs)              // h_refined, 2 MiB
#define NORM_F    (STATS_F + HREF_F)      // norm_dx,   2 MiB

// ------------------------------------------------------------------
// K0: zero the accumulator region (harness does not re-poison)
__global__ void k_zero(float* ws_f) {
    if (threadIdx.x < STATS_F) ws_f[threadIdx.x] = 0.0f;
}

// ------------------------------------------------------------------
// K1: tap moments of zero-padded h_map: E[s_t], E[s_t s_u] (t<=u)
__global__ void k_stats(const float* __restrict__ hmap, float* ws_f) {
    __shared__ float lacc[54];
    if (threadIdx.x < 54) lacc[threadIdx.x] = 0.0f;
    __syncthreads();

    float aS[9];  float aP[45];
#pragma unroll
    for (int t = 0; t < 9; ++t) aS[t] = 0.0f;
#pragma unroll
    for (int i = 0; i < 45; ++i) aP[i] = 0.0f;

    int gt = blockIdx.x * blockDim.x + threadIdx.x;
    int stride = gridDim.x * blockDim.x;
    for (int p = gt; p < NPIX; p += stride) {
        int b = p >> 16, rem = p & (HWs - 1);
        int y = rem >> 8, x = rem & 255;
        float s[9];
#pragma unroll
        for (int e = 0; e < 9; ++e) {
            int yy = y + e / 3 - 1, xx = x + e % 3 - 1;
            s[e] = (yy >= 0 && yy < Hh && xx >= 0 && xx < Ww)
                       ? hmap[b * HWs + yy * Ww + xx] : 0.0f;
        }
#pragma unroll
        for (int t = 0; t < 9; ++t) aS[t] += s[t];
        int idx = 0;
#pragma unroll
        for (int t = 0; t < 9; ++t)
#pragma unroll
            for (int u = t; u < 9; ++u) { aP[idx] += s[t] * s[u]; ++idx; }
    }
#pragma unroll
    for (int t = 0; t < 9; ++t)  atomicAdd(&lacc[t], aS[t]);
#pragma unroll
    for (int i = 0; i < 45; ++i) atomicAdd(&lacc[9 + i], aP[i]);
    __syncthreads();
    if (threadIdx.x < 54) atomicAdd(&ws_f[threadIdx.x], lacc[threadIdx.x]);
}

// ------------------------------------------------------------------
// K2: analytic training-mode BN params from the linear conv:
//     h_c = sum_t w_ct s_t + b_c ; mean/var over (B,H,W) from moments.
__global__ void k_bnparams(const float* __restrict__ w1, const float* __restrict__ b1,
                           const float* __restrict__ gamma, const float* __restrict__ beta,
                           float* ws_f) {
    int c = threadIdx.x;
    if (c >= 16) return;
    const float invN = 1.0f / (float)NPIX;
    float M[9];
#pragma unroll
    for (int t = 0; t < 9; ++t) M[t] = ws_f[t] * invN;
    float E[9][9];
    int idx = 0;
#pragma unroll
    for (int t = 0; t < 9; ++t)
#pragma unroll
        for (int u = t; u < 9; ++u) {
            float v = ws_f[9 + idx] * invN;
            E[t][u] = v; E[u][t] = v; ++idx;
        }
    float wv[9];
#pragma unroll
    for (int t = 0; t < 9; ++t) wv[t] = w1[c * 9 + t];
    float m0 = 0.0f;
#pragma unroll
    for (int t = 0; t < 9; ++t) m0 += wv[t] * M[t];
    float evv = 0.0f;
#pragma unroll
    for (int t = 0; t < 9; ++t)
#pragma unroll
        for (int u = 0; u < 9; ++u) evv += wv[t] * wv[u] * E[t][u];
    float var   = evv - m0 * m0;                   // bias cancels
    float scale = gamma[c] * rsqrtf(var + 1e-5f);
    float shift = beta[c] - m0 * scale;            // applied to (D = conv w/o bias)
    ws_f[SCALE_I + c] = scale;
    ws_f[SHIFT_I + c] = shift;
}

// ------------------------------------------------------------------
// K3: fused conv1(3x3,1->16) + BN + ReLU + conv2(1x1,16->1) via WMMA.
// One wave per 16-pixel tile. GEMM: D[M=ch16, N=px16] = A(16x32 f16) x B(32x16 f16).
// A layout (ISA 16-bit A 16x32): lane m: M=m%16; halves e<8 -> K {e | e+8},
// e>=8 -> K {e+8 | e+16} depending on lane half.
// B layout: same K order as A; lanes 0-15 col N=lane hold K=0..7,16..23; lanes 16-31 hold K=8..15,24..31 (K>=9 zero).
__global__ void k_conv(const float* __restrict__ hmap, const float* __restrict__ w1,
                       const float* __restrict__ w2, const float* __restrict__ b2,
                       float* ws_f) {
    __shared__ float sc[16], sh[16];
    if (threadIdx.x < 16) {
        sc[threadIdx.x] = ws_f[SCALE_I + threadIdx.x];
        sh[threadIdx.x] = ws_f[SHIFT_I + threadIdx.x];
    }
    __syncthreads();

    int wv   = (blockIdx.x * blockDim.x + threadIdx.x) >> 5;
    int lane = threadIdx.x & 31;
    int b   = wv >> 12;               // / (256*16)
    int rem = wv & 4095;
    int y   = rem >> 4;
    int x0  = (rem & 15) << 4;
    int c   = lane & 15;              // A-matrix row (out channel)

    v16h amat, bmat;
#pragma unroll
    for (int e = 0; e < 16; ++e) {
        int K = (lane < 16) ? ((e < 8) ? e : e + 8)
                            : ((e < 8) ? e + 8 : e + 16);
        amat[e] = (K < 9) ? (_Float16)w1[c * 9 + K] : (_Float16)0.0f;
    }
#pragma unroll
    for (int e = 0; e < 16; ++e) {
        float v = 0.0f;
        int K = (lane < 16) ? ((e < 8) ? e : e + 8)
                            : ((e < 8) ? e + 8 : e + 16);
        if (K < 9) {                            // K tap (table order) for pixel N=lane%16
            int yy = y + K / 3 - 1;
            int xx = x0 + (lane & 15) + K % 3 - 1;
            if (yy >= 0 && yy < Hh && xx >= 0 && xx < Ww)
                v = hmap[b * HWs + yy * Ww + xx];
        }
        bmat[e] = (_Float16)v;
    }

    v8f acc = {};
    acc = __builtin_amdgcn_wmma_f32_16x16x32_f16(
        /*neg_a=*/false, amat, /*neg_b=*/false, bmat,
        /*c_mod=*/(short)0, acc, /*reuse_a=*/false, /*reuse_b=*/false);

    // D layout: lane holds pixel N=lane%16, channels M = j + (lane<16 ? 0 : 8)
    int cbase = (lane < 16) ? 0 : 8;
    float partial = 0.0f;
#pragma unroll
    for (int j = 0; j < 8; ++j) {
        int cc = cbase + j;
        float hv = acc[j] * sc[cc] + sh[cc];
        hv = hv > 0.0f ? hv : 0.0f;             // ReLU
        partial += w2[cc] * hv;                 // 1x1 conv partial dot
    }
    float other = __shfl_xor(partial, 16, 32);  // combine channel halves
    if (lane < 16) {
        float* hr = ws_f + STATS_F;
        hr[b * HWs + y * Ww + (x0 + lane)] = partial + other + b2[0];
    }
}

// ------------------------------------------------------------------
// K4: norm_dx per row + global sum. Wave-per-row, 16-row chunks so each
// x row is read ~1.06x (prev kept in registers). lane covers 4 floats.
__global__ void k_normdx(const float* __restrict__ x, float* ws_f) {
    float* norm_dx = ws_f + NORM_F;
    int wv   = (blockIdx.x * blockDim.x + threadIdx.x) >> 5;
    int lane = threadIdx.x & 31;
    int r0   = wv * 16;
    int l0   = r0 & (Ls - 1);

    float4 prev;
    if (l0 == 0) { prev.x = prev.y = prev.z = prev.w = 0.0f; }  // dx[0] = x[0]
    else prev = ((const float4*)(x + (size_t)(r0 - 1) * Dd))[lane];

    float lsum = 0.0f;
    for (int i = 0; i < 16; ++i) {
        int r = r0 + i;
        const float4 cur = ((const float4*)(x + (size_t)r * Dd))[lane];
        __builtin_prefetch((const char*)(x + (size_t)(r + 4) * Dd) + lane * 16, 0, 0);
        float dx0 = cur.x - prev.x, dx1 = cur.y - prev.y;
        float dx2 = cur.z - prev.z, dx3 = cur.w - prev.w;
        float ss = dx0 * dx0 + dx1 * dx1 + dx2 * dx2 + dx3 * dx3;
#pragma unroll
        for (int m = 16; m >= 1; m >>= 1) ss += __shfl_xor(ss, m, 32);
        if (lane == 0) {
            float nv = sqrtf(ss);
            norm_dx[r] = nv;
            lsum += nv;
        }
        prev = cur;
    }
    if (lane == 0) atomicAdd(&ws_f[NSUM_I], lsum);
}

// ------------------------------------------------------------------
// K5: gate + multiply. dir_idx==4 diagonal scan: l=(w,h)-major, value
// h_refined[b, h, (h+w)%W]. Wave-per-row streaming of x -> out.
__global__ void k_gate(const float* __restrict__ x, const float* __restrict__ alpha_p,
                       float* __restrict__ out, float* ws_f) {
    const float* hr      = ws_f + STATS_F;
    const float* norm_dx = ws_f + NORM_F;
    int wv   = (blockIdx.x * blockDim.x + threadIdx.x) >> 5;
    int lane = threadIdx.x & 31;
    int r0   = wv * 16;
    float coef = alpha_p[0] * (ws_f[NSUM_I] * (1.0f / (float)NROW) + 1e-8f);

    for (int i = 0; i < 16; ++i) {
        int r = r0 + i;
        float gate = 0.0f;
        if (lane == 0) {
            int b = r >> 16, l = r & (Ls - 1);
            float nv = norm_dx[r];
            int w_ = l >> 8, hh = l & 255;
            float scur = hr[b * HWs + hh * Ww + ((hh + w_) & 255)];
            float dh;
            if (l == 0) dh = fabsf(scur);
            else {
                int lp = l - 1, wp = lp >> 8, hp = lp & 255;
                float sp = hr[b * HWs + hp * Ww + ((hp + wp) & 255)];
                dh = fabsf(scur - sp);
            }
            float gs = nv + coef * dh;
            gate = 0.2f + 0.8f * tanhf(gs);
        }
        gate = __shfl(gate, 0, 32);
        const float4 cur = ((const float4*)(x + (size_t)r * Dd))[lane];
        __builtin_prefetch((const char*)(x + (size_t)(r + 4) * Dd) + lane * 16, 0, 0);
        float4 o;
        o.x = cur.x * gate; o.y = cur.y * gate;
        o.z = cur.z * gate; o.w = cur.w * gate;
        ((float4*)(out + (size_t)r * Dd))[lane] = o;
    }
}

// ------------------------------------------------------------------
extern "C" void kernel_launch(void* const* d_in, const int* in_sizes, int n_in,
                              void* d_out, int out_size, void* d_ws, size_t ws_size,
                              hipStream_t stream) {
    const float* x     = (const float*)d_in[0];
    const float* hmap  = (const float*)d_in[1];
    const float* w1    = (const float*)d_in[2];
    const float* b1    = (const float*)d_in[3];
    const float* gamma = (const float*)d_in[4];
    const float* beta  = (const float*)d_in[5];
    const float* w2    = (const float*)d_in[6];
    const float* b2    = (const float*)d_in[7];
    const float* alpha = (const float*)d_in[8];
    // d_in[9..11] = dir_idx(=4), H(=256), W(=256): specialized at compile time.
    float* out  = (float*)d_out;
    float* ws_f = (float*)d_ws;

    k_zero    <<<1,    256, 0, stream>>>(ws_f);
    k_stats   <<<128,  256, 0, stream>>>(hmap, ws_f);
    k_bnparams<<<1,    32,  0, stream>>>(w1, b1, gamma, beta, ws_f);
    k_conv    <<<4096, 256, 0, stream>>>(hmap, w1, w2, b2, ws_f);   // 32768 waves = tiles
    k_normdx  <<<4096, 256, 0, stream>>>(x, ws_f);                  // 32768 waves * 16 rows
    k_gate    <<<4096, 256, 0, stream>>>(x, alpha, out, ws_f);
}
